// EdgeBlock_17703855194353
// MI455X (gfx1250) — matrix operand order repaired and checked by reading, not compiled, not run on hardware
//
#include <hip/hip_runtime.h>

// ---------------------------------------------------------------------------
// Types for CDNA5 WMMA (wave32, bf16 in / f32 accumulate)
// ---------------------------------------------------------------------------
typedef __bf16 bf16_t;
typedef __attribute__((ext_vector_type(16))) __bf16 v16bf;
typedef __attribute__((ext_vector_type(8)))  __bf16 v8bf;
typedef __attribute__((ext_vector_type(4)))  __bf16 v4bf;
typedef __attribute__((ext_vector_type(8)))  float  v8f;

#define D_FEAT   128
#define K_FEAT   384        // 3*D
#define TPB_MAIN 512        // 16 waves
#define BLOCK_E  128        // edges per workgroup

// LDS strides (bf16 elements), padded +8 to break bank conflicts
#define W1_STRIDE 392       // 384 + 8
#define W2_STRIDE 136       // 128 + 8
#define X_STRIDE  392
#define H_STRIDE  136
#define SW1_ELEMS (128 * W1_STRIDE)
#define SW2_ELEMS (128 * W2_STRIDE)
#define SX_ELEMS  (BLOCK_E * X_STRIDE)
#define SH_ELEMS  (BLOCK_E * H_STRIDE)
#define SMEM_BYTES ((SW1_ELEMS + SW2_ELEMS + SX_ELEMS + SH_ELEMS) * 2)

static __device__ __forceinline__ bf16_t f2bf(float f) { return (bf16_t)f; }

// ---------------------------------------------------------------------------
// Kernel 1: cell_attr[c][d] = sum of 3 gathered node rows, stored as bf16.
// One thread per (cell, 4-float chunk): C*32 threads.
// ---------------------------------------------------------------------------
__global__ __launch_bounds__(256)
void cell_reduce_kernel(const float* __restrict__ node_attr,
                        const int*   __restrict__ cells_node,
                        bf16_t*      __restrict__ cellb,
                        int C_) {
    int t = blockIdx.x * blockDim.x + threadIdx.x;
    int c = t >> 5;
    int j = t & 31;
    if (c >= C_) return;
    int i0 = cells_node[3 * c + 0];
    int i1 = cells_node[3 * c + 1];
    int i2 = cells_node[3 * c + 2];
    const float4 a = *(const float4*)&node_attr[(long)i0 * D_FEAT + j * 4];
    const float4 b = *(const float4*)&node_attr[(long)i1 * D_FEAT + j * 4];
    const float4 d = *(const float4*)&node_attr[(long)i2 * D_FEAT + j * 4];
    v4bf o;
    o[0] = f2bf(a.x + b.x + d.x);
    o[1] = f2bf(a.y + b.y + d.y);
    o[2] = f2bf(a.z + b.z + d.z);
    o[3] = f2bf(a.w + b.w + d.w);
    *(v4bf*)&cellb[(long)c * D_FEAT + j * 4] = o;
}

// ---------------------------------------------------------------------------
// Kernel 2: transpose fp32 W[K][Nn] -> bf16 Wt[Nn][K] (column-major weights so
// B fragments are contiguous-in-K per output channel).
// ---------------------------------------------------------------------------
__global__ __launch_bounds__(256)
void transpose_bf16_kernel(const float* __restrict__ W,
                           bf16_t*      __restrict__ Wt,
                           int K, int Nn) {
    int t = blockIdx.x * blockDim.x + threadIdx.x;
    if (t >= K * Nn) return;
    int k = t / Nn;
    int n = t % Nn;
    Wt[(long)n * K + k] = f2bf(W[(long)k * Nn + n]);
}

// ---------------------------------------------------------------------------
// Kernel 3: fused edge MLP.
//   X = [cell_bf16[s] | cell_bf16[r]*(s!=r) | bf16(edge_attr)]  (128 x 384)
//   H = relu(X @ W1 + b1)   via v_wmma_f32_16x16x32_bf16, K=384 -> 12 steps
//   O = H @ W2 + b2         via v_wmma_f32_16x16x32_bf16, K=128 -> 4 steps
//
// 16 waves; wave w owns a 2x2 grid of 16x16 tiles:
//   col tiles ct0 = (w&3)*2, ct0+1 ; row tiles rt0 = (w>>2)*2, rt0+1
//
// Fragment layouts (CDNA5 ISA 7.12.2, wave32):
//   A 16x32 bf16 : lane l -> row m=l&15; elems i<8: K=kb+i, i>=8: K=16+kb+i-8,
//                  kb=(l>>4)*8  -> two 16B LDS reads per fragment
//   B 32x16 bf16 : lane l -> col n=l&15; same K mapping as A
//   C/D 16x16 f32: lane l -> col n=l&15; VGPR v -> row m=v+(l>>4)*8
// ---------------------------------------------------------------------------
__global__ __launch_bounds__(TPB_MAIN)
void edge_mlp_kernel(const bf16_t* __restrict__ cellb,   // [C][128] bf16
                     const bf16_t* __restrict__ w1t,     // [128][384] bf16
                     const bf16_t* __restrict__ w2t,     // [128][128] bf16
                     const float*  __restrict__ b1,
                     const float*  __restrict__ b2,
                     const float*  __restrict__ edge_attr, // [E][128] f32
                     const int*    __restrict__ senders,
                     const int*    __restrict__ receivers,
                     float*        __restrict__ out,       // [E][128] f32
                     int E_) {
    extern __shared__ __align__(16) char smem[];
    bf16_t* sW1 = (bf16_t*)smem;            // 128 x W1_STRIDE
    bf16_t* sW2 = sW1 + SW1_ELEMS;          // 128 x W2_STRIDE
    bf16_t* sX  = sW2 + SW2_ELEMS;          // BLOCK_E x X_STRIDE
    bf16_t* sH  = sX  + SX_ELEMS;           // BLOCK_E x H_STRIDE

    const int tid = threadIdx.x;

    // ---- stage W1^T, W2^T into LDS (16B vector copies) ----
    for (int c = tid; c < 128 * (K_FEAT / 8); c += TPB_MAIN) {   // 6144 chunks
        int n = c / (K_FEAT / 8);
        int off = (c % (K_FEAT / 8)) * 8;
        *(v8bf*)&sW1[n * W1_STRIDE + off] = *(const v8bf*)&w1t[(long)n * K_FEAT + off];
    }
    for (int c = tid; c < 128 * (D_FEAT / 8); c += TPB_MAIN) {   // 2048 chunks
        int n = c / (D_FEAT / 8);
        int off = (c % (D_FEAT / 8)) * 8;
        *(v8bf*)&sW2[n * W2_STRIDE + off] = *(const v8bf*)&w2t[(long)n * D_FEAT + off];
    }

    // ---- gather X tile: thread = (edge-in-block, quarter of 32 features) ----
    const int e_base = blockIdx.x * BLOCK_E;
    {
        int le = tid >> 2;       // 0..127 edge within block
        int q  = tid & 3;        // 32-feature quarter
        int e  = e_base + le;
        bool valid = (e < E_);
        int s = 0, r = 0;
        if (valid) { s = senders[e]; r = receivers[e]; }
        bf16_t* xrow = &sX[le * X_STRIDE];
        v8bf z = {};
        // sender block [0,128)
        {
            const v8bf* src = (const v8bf*)&cellb[(long)s * D_FEAT + q * 32];
            #pragma unroll
            for (int i = 0; i < 4; ++i) {
                v8bf v;
                if (valid) v = src[i]; else v = z;
                *(v8bf*)&xrow[q * 32 + i * 8] = v;
            }
        }
        // receiver block [128,256), masked for self-edges
        {
            bool keep = valid && (s != r);
            const v8bf* src = (const v8bf*)&cellb[(long)r * D_FEAT + q * 32];
            #pragma unroll
            for (int i = 0; i < 4; ++i) {
                v8bf v;
                if (keep) v = src[i]; else v = z;
                *(v8bf*)&xrow[128 + q * 32 + i * 8] = v;
            }
        }
        // edge_attr block [256,384), fp32 -> bf16
        {
            const float4* src = (const float4*)&edge_attr[(long)e * D_FEAT + q * 32];
            #pragma unroll
            for (int i = 0; i < 8; ++i) {
                float4 f;
                if (valid) f = src[i]; else { f.x = f.y = f.z = f.w = 0.f; }
                v4bf o;
                o[0] = f2bf(f.x); o[1] = f2bf(f.y); o[2] = f2bf(f.z); o[3] = f2bf(f.w);
                *(v4bf*)&xrow[256 + q * 32 + i * 4] = o;
            }
        }
    }
    __syncthreads();

    // ---- tile assignment ----
    const int wave = tid >> 5;
    const int lane = tid & 31;
    const int nw   = lane & 15;     // column within tile / A-row within tile
    const int hk   = lane >> 4;     // half-wave: K-offset selector
    const int ct0  = (wave & 3) * 2;        // 2 col tiles
    const int rt0  = (wave >> 2) * 2;       // 2 row tiles
    const int ncol0 = ct0 * 16 + nw;
    const int ncol1 = ncol0 + 16;

    v8f acc[2][2];
    v8f zacc = {};
    acc[0][0] = zacc; acc[0][1] = zacc; acc[1][0] = zacc; acc[1][1] = zacc;

    // ---- GEMM1: H = relu(X @ W1 + b1), K=384 ----
    {
        const bf16_t* w1a = &sW1[ncol0 * W1_STRIDE + hk * 8];
        const bf16_t* w1b = &sW1[ncol1 * W1_STRIDE + hk * 8];
        const bf16_t* xr0 = &sX[(rt0 * 16 + nw) * X_STRIDE + hk * 8];
        const bf16_t* xr1 = xr0 + 16 * X_STRIDE;
        #pragma unroll
        for (int kk = 0; kk < K_FEAT / 32; ++kk) {
            const int k0 = kk * 32;
            v16bf bf0, bf1, af0, af1;
            ((v8bf*)&bf0)[0] = *(const v8bf*)&w1a[k0];
            ((v8bf*)&bf0)[1] = *(const v8bf*)&w1a[k0 + 16];
            ((v8bf*)&bf1)[0] = *(const v8bf*)&w1b[k0];
            ((v8bf*)&bf1)[1] = *(const v8bf*)&w1b[k0 + 16];
            ((v8bf*)&af0)[0] = *(const v8bf*)&xr0[k0];
            ((v8bf*)&af0)[1] = *(const v8bf*)&xr0[k0 + 16];
            ((v8bf*)&af1)[0] = *(const v8bf*)&xr1[k0];
            ((v8bf*)&af1)[1] = *(const v8bf*)&xr1[k0 + 16];
            acc[0][0] = __builtin_amdgcn_wmma_f32_16x16x32_bf16(false, af0, false, bf0, (short)0, acc[0][0], false, false);
            acc[0][1] = __builtin_amdgcn_wmma_f32_16x16x32_bf16(false, af0, false, bf1, (short)0, acc[0][1], false, false);
            acc[1][0] = __builtin_amdgcn_wmma_f32_16x16x32_bf16(false, af1, false, bf0, (short)0, acc[1][0], false, false);
            acc[1][1] = __builtin_amdgcn_wmma_f32_16x16x32_bf16(false, af1, false, bf1, (short)0, acc[1][1], false, false);
        }
    }
    // bias + relu -> bf16 H tile in LDS
    {
        const float bias[2] = { b1[ncol0], b1[ncol1] };
        #pragma unroll
        for (int i = 0; i < 2; ++i) {           // row tile
            #pragma unroll
            for (int j = 0; j < 2; ++j) {       // col tile
                const int nc = (j == 0) ? ncol0 : ncol1;
                #pragma unroll
                for (int v = 0; v < 8; ++v) {
                    float h = acc[i][j][v] + bias[j];
                    h = h > 0.f ? h : 0.f;
                    int m = (rt0 + i) * 16 + hk * 8 + v;
                    sH[m * H_STRIDE + nc] = f2bf(h);
                }
            }
        }
    }
    __syncthreads();

    // ---- GEMM2: O = H @ W2 + b2, K=128 ----
    acc[0][0] = zacc; acc[0][1] = zacc; acc[1][0] = zacc; acc[1][1] = zacc;
    {
        const bf16_t* w2a = &sW2[ncol0 * W2_STRIDE + hk * 8];
        const bf16_t* w2b = &sW2[ncol1 * W2_STRIDE + hk * 8];
        const bf16_t* hr0 = &sH[(rt0 * 16 + nw) * H_STRIDE + hk * 8];
        const bf16_t* hr1 = hr0 + 16 * H_STRIDE;
        #pragma unroll
        for (int kk = 0; kk < D_FEAT / 32; ++kk) {
            const int k0 = kk * 32;
            v16bf bf0, bf1, af0, af1;
            ((v8bf*)&bf0)[0] = *(const v8bf*)&w2a[k0];
            ((v8bf*)&bf0)[1] = *(const v8bf*)&w2a[k0 + 16];
            ((v8bf*)&bf1)[0] = *(const v8bf*)&w2b[k0];
            ((v8bf*)&bf1)[1] = *(const v8bf*)&w2b[k0 + 16];
            ((v8bf*)&af0)[0] = *(const v8bf*)&hr0[k0];
            ((v8bf*)&af0)[1] = *(const v8bf*)&hr0[k0 + 16];
            ((v8bf*)&af1)[0] = *(const v8bf*)&hr1[k0];
            ((v8bf*)&af1)[1] = *(const v8bf*)&hr1[k0 + 16];
            acc[0][0] = __builtin_amdgcn_wmma_f32_16x16x32_bf16(false, af0, false, bf0, (short)0, acc[0][0], false, false);
            acc[0][1] = __builtin_amdgcn_wmma_f32_16x16x32_bf16(false, af0, false, bf1, (short)0, acc[0][1], false, false);
            acc[1][0] = __builtin_amdgcn_wmma_f32_16x16x32_bf16(false, af1, false, bf0, (short)0, acc[1][0], false, false);
            acc[1][1] = __builtin_amdgcn_wmma_f32_16x16x32_bf16(false, af1, false, bf1, (short)0, acc[1][1], false, false);
        }
    }
    // bias + store fp32 output (lanes 0..15 -> 64B contiguous per row segment)
    {
        const float bias[2] = { b2[ncol0], b2[ncol1] };
        #pragma unroll
        for (int i = 0; i < 2; ++i) {
            #pragma unroll
            for (int v = 0; v < 8; ++v) {
                int e = e_base + (rt0 + i) * 16 + hk * 8 + v;
                if (e < E_) {
                    out[(long)e * D_FEAT + ncol0] = acc[i][0][v] + bias[0];
                    out[(long)e * D_FEAT + ncol1] = acc[i][1][v] + bias[1];
                }
            }
        }
    }
}

// ---------------------------------------------------------------------------
// Launch
// Inputs: 0 node_attr[N,128] f32, 1 edge_attr[E,128] f32, 2 W1[384,128] f32,
//         3 b1[128] f32, 4 W2[128,128] f32, 5 b2[128] f32,
//         6 cells_node[3C] i32, 7 cells_index[3C] i32, 8 edge_index[2,E] i32
// Output: edge_attr_new[E,128] f32
// ---------------------------------------------------------------------------
extern "C" void kernel_launch(void* const* d_in, const int* in_sizes, int n_in,
                              void* d_out, int out_size, void* d_ws, size_t ws_size,
                              hipStream_t stream) {
    const float* node_attr = (const float*)d_in[0];
    const float* edge_attr = (const float*)d_in[1];
    const float* W1        = (const float*)d_in[2];
    const float* b1        = (const float*)d_in[3];
    const float* W2        = (const float*)d_in[4];
    const float* b2        = (const float*)d_in[5];
    const int*   cells_node = (const int*)d_in[6];
    const int*   edge_index = (const int*)d_in[8];
    float* out = (float*)d_out;

    const int C = in_sizes[7] / 3;       // cells_index = repeat(arange(C), 3)
    const int E = in_sizes[1] / D_FEAT;
    const int* senders   = edge_index;
    const int* receivers = edge_index + E;

    // workspace carve: cell_attr bf16 | W1^T bf16 | W2^T bf16
    char* ws = (char*)d_ws;
    bf16_t* cellb = (bf16_t*)ws;
    size_t off = ((size_t)C * D_FEAT * sizeof(bf16_t) + 255) & ~(size_t)255;
    bf16_t* w1t = (bf16_t*)(ws + off);
    off += (size_t)K_FEAT * D_FEAT * sizeof(bf16_t);
    off = (off + 255) & ~(size_t)255;
    bf16_t* w2t = (bf16_t*)(ws + off);

    // 1) cell_attr = fan-in-3 scatter-sum, bf16
    {
        long threads = (long)C * 32;
        int grid = (int)((threads + 255) / 256);
        cell_reduce_kernel<<<grid, 256, 0, stream>>>(node_attr, cells_node, cellb, C);
    }
    // 2) weight transposes to bf16
    transpose_bf16_kernel<<<(K_FEAT * D_FEAT + 255) / 256, 256, 0, stream>>>(W1, w1t, K_FEAT, D_FEAT);
    transpose_bf16_kernel<<<(D_FEAT * D_FEAT + 255) / 256, 256, 0, stream>>>(W2, w2t, D_FEAT, D_FEAT);
    // 3) fused edge MLP (WMMA)
    {
        int grid = (E + BLOCK_E - 1) / BLOCK_E;
        edge_mlp_kernel<<<grid, TPB_MAIN, SMEM_BYTES, stream>>>(
            cellb, w1t, w2t, b1, b2, edge_attr, senders, receivers, out, E);
    }
}
